// non_local_layer_36335423324632
// MI455X (gfx1250) — matrix operand order repaired and checked by reading, not compiled, not run on hardware
//
#include <hip/hip_runtime.h>
#include <hip/hip_bf16.h>
#include <cstddef>

// ---------------------------------------------------------------------------
// Non-local attention block, MI455X (gfx1250, wave32, WMMA bf16 K=32 path).
// B=8, C=1024, Cb=512, N=H*W=2304.  ~169 GFLOP of GEMM -> all on v_wmma.
// Compile-time leading dims -> immediate-offset stores; 32x64 wave tiles.
// ---------------------------------------------------------------------------

typedef __attribute__((ext_vector_type(16))) __bf16        bhalf16;
typedef __attribute__((ext_vector_type(8)))  __bf16        bhalf8;
typedef __attribute__((ext_vector_type(4)))  __bf16        bhalf4;
typedef __attribute__((ext_vector_type(8)))  float         floatx8;
typedef __attribute__((ext_vector_type(4)))  float         floatx4;
typedef __attribute__((ext_vector_type(4)))  unsigned int  uint4v;
typedef __attribute__((ext_vector_type(8)))  unsigned int  uint8v;

// ------------------------------- converts ----------------------------------
__global__ __launch_bounds__(256)
void f32_to_bf16_kernel(const float* __restrict__ src, __bf16* __restrict__ dst, size_t n4) {
    size_t i = (size_t)blockIdx.x * blockDim.x + threadIdx.x;
    if (i < n4) {
        floatx4 f = *(const floatx4*)(src + i * 4);
        bhalf4 h;
        h[0] = (__bf16)f[0]; h[1] = (__bf16)f[1];
        h[2] = (__bf16)f[2]; h[3] = (__bf16)f[3];
        *(bhalf4*)(dst + i * 4) = h;
    }
}

// ------------------------------- WMMA GEMM ---------------------------------
// D[M x N] = alpha * A[M x K] * B[K x N]
//   TA: A stored column-major (A(m,k) = Aptr[k*LDA + m])
//   TB: B stored as [N][K]    (B(k,n) = Bptr[n*LDB + k])
// Block: 128 threads = 4 waves, 64x128 tile; each wave 32x64 via 2x4 WMMA.
constexpr int BM = 64, BN = 128, BK = 32;
constexpr int LDA_S = 40;   // bf16 elems; 80 B row stride: 16B-aligned, padded
constexpr int LDBT_S = 40;  // sBt is [n][k]

template<bool TA, bool TB, bool OUT_BF16, int LDA, int LDB, int LDD>
__global__ __launch_bounds__(128)
void gemm_wmma_bf16(const __bf16* __restrict__ A,
                    const __bf16* __restrict__ B,
                    void* __restrict__ D,
                    int K, float alpha)
{
    __shared__ alignas(16) __bf16 sA [BM * LDA_S];   // [m][k]
    __shared__ alignas(16) __bf16 sBt[BN * LDBT_S];  // [n][k]  (B transposed)

    const int tid  = threadIdx.x;
    const int lane = tid & 31;
    const int wave = tid >> 5;
    const int wr   = wave >> 1;       // wave row (0..1) -> 32 rows
    const int wc   = wave & 1;        // wave col (0..1) -> 64 cols
    const int hl   = lane >> 4;       // half-wave select
    const int lr   = lane & 15;

    const int m0 = blockIdx.y * BM;
    const int n0 = blockIdx.x * BN;

    floatx8 acc[2][4] = {};

    for (int k0 = 0; k0 < K; k0 += BK) {
        // ---- stage A tile: sA[m][k], 64x32, 256 x 16B chunks -----------
        if (!TA) {
            #pragma unroll
            for (int e = 0; e < 2; ++e) {
                int c = tid + e * 128;
                int m = c >> 2, kc = c & 3;               // k-chunk of 8
                uint4v v = *(const uint4v*)(A + (size_t)(m0 + m) * LDA + k0 + kc * 8);
                *(uint4v*)(&sA[m * LDA_S + kc * 8]) = v;
            }
        } else {
            #pragma unroll
            for (int e = 0; e < 2; ++e) {
                int c = tid + e * 128;
                int k = c >> 3, mc = c & 7;               // m-chunk of 8
                uint4v v = *(const uint4v*)(A + (size_t)(k0 + k) * LDA + m0 + mc * 8);
                bhalf8 hv = __builtin_bit_cast(bhalf8, v);
                #pragma unroll
                for (int t = 0; t < 8; ++t)
                    sA[(mc * 8 + t) * LDA_S + k] = hv[t];
            }
        }
        // ---- stage B tile: sBt[n][k], 128x32, 512 x 16B chunks ---------
        if (TB) {
            #pragma unroll
            for (int e = 0; e < 4; ++e) {
                int c = tid + e * 128;
                int n = c >> 2, kc = c & 3;
                uint4v v = *(const uint4v*)(B + (size_t)(n0 + n) * LDB + k0 + kc * 8);
                *(uint4v*)(&sBt[n * LDBT_S + kc * 8]) = v;
            }
        } else {
            #pragma unroll
            for (int e = 0; e < 4; ++e) {
                int c = tid + e * 128;
                int k = c >> 4, nc = c & 15;              // n-chunk of 8
                uint4v v = *(const uint4v*)(B + (size_t)(k0 + k) * LDB + n0 + nc * 8);
                bhalf8 hv = __builtin_bit_cast(bhalf8, v);
                #pragma unroll
                for (int t = 0; t < 8; ++t)
                    sBt[(nc * 8 + t) * LDBT_S + k] = hv[t];
            }
        }
        __syncthreads();

        // ---- fragments: pure ds_load_b128 ------------------------------
        // A (ISA 16-bit A 16x32): elem j -> K = (j>=8?16:0) + hl*8 + (j&7)
        //   = two contiguous 16B chunks at k = hl*8 and k = 16 + hl*8.
        // B (32x16):              elem j -> K = (j>=8?16:0) + hl*8 + (j&7)
        //   = two contiguous 16B chunks in sBt at k = hl*8 and k = 16 + hl*8.
        bhalf16 afrag[2], bfrag[4];
        #pragma unroll
        for (int t = 0; t < 2; ++t) {
            const __bf16* ap = &sA[(wr * 32 + t * 16 + lr) * LDA_S + hl * 8];
            uint4v alo = *(const uint4v*)ap;
            uint4v ahi = *(const uint4v*)(ap + 16);
            afrag[t] = __builtin_bit_cast(bhalf16,
                (uint8v)__builtin_shufflevector(alo, ahi, 0, 1, 2, 3, 4, 5, 6, 7));
        }
        #pragma unroll
        for (int t = 0; t < 4; ++t) {
            const __bf16* bp = &sBt[(wc * 64 + t * 16 + lr) * LDBT_S + hl * 8];
            uint4v blo = *(const uint4v*)bp;
            uint4v bhi = *(const uint4v*)(bp + 16);
            bfrag[t] = __builtin_bit_cast(bhalf16,
                (uint8v)__builtin_shufflevector(blo, bhi, 0, 1, 2, 3, 4, 5, 6, 7));
        }
        #pragma unroll
        for (int tm = 0; tm < 2; ++tm)
            #pragma unroll
            for (int tn = 0; tn < 4; ++tn)
                acc[tm][tn] = __builtin_amdgcn_wmma_f32_16x16x32_bf16(
                    false, afrag[tm], false, bfrag[tn],
                    (short)0, acc[tm][tn], false, false);
        __syncthreads();
    }

    // store: C/D layout -> VGPR v holds row (v + hl*8), col = lane%16
    #pragma unroll
    for (int tm = 0; tm < 2; ++tm) {
        const int row0 = m0 + wr * 32 + tm * 16 + hl * 8;
        #pragma unroll
        for (int tn = 0; tn < 4; ++tn) {
            const int col = n0 + wc * 64 + tn * 16 + lr;
            if (OUT_BF16) {
                __bf16* dp = (__bf16*)D + (size_t)row0 * LDD + col;
                #pragma unroll
                for (int v = 0; v < 8; ++v)
                    dp[(size_t)v * LDD] = (__bf16)(acc[tm][tn][v] * alpha);
            } else {
                float* dp = (float*)D + (size_t)row0 * LDD + col;
                #pragma unroll
                for (int v = 0; v < 8; ++v)
                    dp[(size_t)v * LDD] = acc[tm][tn][v] * alpha;
            }
        }
    }
}

// ------------------------------- softmax -----------------------------------
__global__ __launch_bounds__(256)
void softmax_row_kernel(float* __restrict__ S, __bf16* __restrict__ attn, int Ncols) {
    __shared__ float red[256];
    const int tid = threadIdx.x;
    float* srow = S + (size_t)blockIdx.x * Ncols;

    float m = -3.4e38f;
    for (int i = tid; i < Ncols; i += 256) m = fmaxf(m, srow[i]);
    red[tid] = m; __syncthreads();
    for (int s = 128; s > 0; s >>= 1) {
        if (tid < s) red[tid] = fmaxf(red[tid], red[tid + s]);
        __syncthreads();
    }
    m = red[0]; __syncthreads();

    float sum = 0.f;
    for (int i = tid; i < Ncols; i += 256) {
        float e = __expf(srow[i] - m);
        srow[i] = e;
        sum += e;
    }
    red[tid] = sum; __syncthreads();
    for (int s = 128; s > 0; s >>= 1) {
        if (tid < s) red[tid] += red[tid + s];
        __syncthreads();
    }
    const float inv = 1.0f / red[0];

    __bf16* arow = attn + (size_t)blockIdx.x * Ncols;
    for (int i = tid; i < Ncols; i += 256) arow[i] = (__bf16)(srow[i] * inv);
}

// ------------------------------- batchnorm ---------------------------------
__global__ __launch_bounds__(256)
void bn_stats_kernel(const float* __restrict__ wy, float* __restrict__ mean,
                     float* __restrict__ inv, int Bn, int C, int Nn) {
    __shared__ float rs[256], rq[256];
    const int c = blockIdx.x;
    float s = 0.f, q = 0.f;
    for (int b = 0; b < Bn; ++b) {
        const float* p = wy + ((size_t)b * C + c) * Nn;
        for (int i = threadIdx.x; i < Nn; i += 256) {
            float v = p[i]; s += v; q += v * v;
        }
    }
    rs[threadIdx.x] = s; rq[threadIdx.x] = q; __syncthreads();
    for (int st = 128; st > 0; st >>= 1) {
        if (threadIdx.x < st) {
            rs[threadIdx.x] += rs[threadIdx.x + st];
            rq[threadIdx.x] += rq[threadIdx.x + st];
        }
        __syncthreads();
    }
    if (threadIdx.x == 0) {
        float cnt = (float)Bn * (float)Nn;
        float mu  = rs[0] / cnt;
        float var = rq[0] / cnt - mu * mu;
        mean[c] = mu;
        inv[c]  = rsqrtf(var + 1e-5f);
    }
}

// grid = (N/256, C, B): no integer division in the hot path
__global__ __launch_bounds__(256)
void bn_residual_kernel(float* __restrict__ out, const float* __restrict__ x,
                        const float* __restrict__ mean, const float* __restrict__ inv,
                        const float* __restrict__ gamma, const float* __restrict__ beta,
                        int C, int Nn) {
    const int n = blockIdx.x * 256 + threadIdx.x;
    const int c = blockIdx.y;
    const int b = blockIdx.z;
    size_t i = ((size_t)b * C + c) * Nn + n;
    float v = out[i];
    out[i] = (v - mean[c]) * inv[c] * gamma[c] + beta[c] + x[i];
}

// ------------------------------- launch ------------------------------------
extern "C" void kernel_launch(void* const* d_in, const int* in_sizes, int n_in,
                              void* d_out, int out_size, void* d_ws, size_t ws_size,
                              hipStream_t stream) {
    (void)in_sizes; (void)n_in; (void)out_size; (void)ws_size;
    const float* x     = (const float*)d_in[0];
    const float* Wg    = (const float*)d_in[1];
    const float* Wz    = (const float*)d_in[2];
    const float* gamma = (const float*)d_in[3];
    const float* beta  = (const float*)d_in[4];
    float* out = (float*)d_out;

    constexpr int Bb = 8, C = 1024, Cb = 512, N = 48 * 48;  // 2304

    char* ws = (char*)d_ws;
    size_t off = 0;
    auto alloc = [&](size_t bytes) -> void* {
        void* p = ws + off;
        off += (bytes + 255) & ~(size_t)255;
        return p;
    };
    __bf16* xh   = (__bf16*)alloc((size_t)Bb * C * N * 2); // 37.7 MB
    __bf16* Wgh  = (__bf16*)alloc((size_t)Cb * C * 2);
    __bf16* Wzh  = (__bf16*)alloc((size_t)C * Cb * 2);
    float*  S    = (float*) alloc((size_t)N * N * 4);      // 21.2 MB (per batch)
    __bf16* attn = (__bf16*)alloc((size_t)N * N * 2);      // 10.6 MB (per batch)
    __bf16* gh   = (__bf16*)alloc((size_t)Cb * N * 2);
    __bf16* yh   = (__bf16*)alloc((size_t)Cb * N * 2);
    float*  mean = (float*) alloc((size_t)C * 4);
    float*  inv  = (float*) alloc((size_t)C * 4);

    // fp32 -> bf16 inputs (vectorized x4)
    {
        size_t nx4 = (size_t)Bb * C * N / 4;
        f32_to_bf16_kernel<<<dim3((unsigned)((nx4 + 255) / 256)), 256, 0, stream>>>(x, xh, nx4);
        size_t nw4 = (size_t)Cb * C / 4;
        f32_to_bf16_kernel<<<dim3((unsigned)((nw4 + 255) / 256)), 256, 0, stream>>>(Wg, Wgh, nw4);
        f32_to_bf16_kernel<<<dim3((unsigned)((nw4 + 255) / 256)), 256, 0, stream>>>(Wz, Wzh, nw4);
    }

    for (int b = 0; b < Bb; ++b) {
        const __bf16* xb = xh + (size_t)b * C * N;  // [C][N] bf16
        // 1) S[n,m] = sum_c x[c,n] x[c,m] / N   (A = x^T -> TA, B = x -> normal)
        dim3 gGram(N / BN, N / BM);
        gemm_wmma_bf16<true, false, false, N, N, N><<<gGram, 128, 0, stream>>>(
            xb, xb, (void*)S, C, 1.0f / (float)N);
        // 2) softmax over rows -> bf16 attn
        softmax_row_kernel<<<N, 256, 0, stream>>>(S, attn, N);
        // 3) g = Wg * x : [Cb x C] x [C x N]
        dim3 gG(N / BN, Cb / BM);
        gemm_wmma_bf16<false, false, true, C, N, N><<<gG, 128, 0, stream>>>(
            Wgh, xb, (void*)gh, C, 1.0f);
        // 4) y[d,n] = sum_m g[d,m] attn[n,m] : A = g, B = attn^T (TB)
        gemm_wmma_bf16<false, true, true, N, N, N><<<gG, 128, 0, stream>>>(
            gh, attn, (void*)yh, N, 1.0f);
        // 5) wy = Wz * y : [C x Cb] x [Cb x N] -> fp32 straight into d_out
        dim3 gW(N / BN, C / BM);
        gemm_wmma_bf16<false, false, false, Cb, N, N><<<gW, 128, 0, stream>>>(
            Wzh, yh, (void*)(out + (size_t)b * C * N), Cb, 1.0f);
    }

    // BatchNorm statistics over (B, N) per channel, then fused norm + residual
    bn_stats_kernel<<<C, 256, 0, stream>>>(out, mean, inv, Bb, C, N);
    bn_residual_kernel<<<dim3(N / 256, C, Bb), 256, 0, stream>>>(
        out, x, mean, inv, gamma, beta, C, N);
}
